// HeterogeneousGAT_45045617001042
// MI455X (gfx1250) — matrix operand order repaired and checked by reading, not compiled, not run on hardware
//
#include <hip/hip_runtime.h>
#include <hip/hip_bf16.h>

#define NUSER 100000
#define NTX   300000
#define NMERCH 10000

typedef __attribute__((ext_vector_type(16))) __bf16 v16bf;
typedef __attribute__((ext_vector_type(8)))  float  v8f;

union ABits { uint4 q[2]; v16bf v; };

__device__ __forceinline__ unsigned short f2bf(float f) {
  unsigned u = __float_as_uint(f);
  u += 0x7FFFu + ((u >> 16) & 1u);   // round-to-nearest-even
  return (unsigned short)(u >> 16);
}

// monotone float <-> uint encoding for atomicMax on signed floats
__device__ __forceinline__ unsigned fenc(float f) {
  unsigned u = __float_as_uint(f);
  return u ^ (unsigned)(((int)u >> 31) | 0x80000000);
}
__device__ __forceinline__ float fdec(unsigned k) {
  unsigned m = ((int)k < 0) ? 0x80000000u : 0xFFFFFFFFu;
  return __uint_as_float(k ^ m);
}

// ---- weight f32[K,64] -> bf16 swizzled WMMA-B layout: [(kt*4+nt)*32+lane]*16+e ----
// lane n-col = nt*16+(lane&15); k = kt*32 + (lane>=16?8:0) + (e&7) + (e>=8?16:0)
__global__ void k_swizzle_w(const float* __restrict__ W, unsigned short* __restrict__ out,
                            int Ksrc, int Kpad) {
  int t = blockIdx.x * blockDim.x + threadIdx.x;
  if (t >= Kpad * 64) return;
  int e    = t & 15;
  int lane = (t >> 4) & 31;
  int nt   = (t >> 9) & 3;
  int kt   = t >> 11;
  int k = kt * 32 + ((lane >> 4) << 3) + (e & 7) + ((e >> 3) << 4);
  int n = nt * 16 + (lane & 15);
  float v = (k < Ksrc) ? W[k * 64 + n] : 0.0f;
  out[t] = f2bf(v);
}

// ---- f32[rows,sc] -> bf16[rows,dc] with zero pad (dc >= sc) ----
__global__ void k_f32_to_bf16(const float* __restrict__ src, unsigned short* __restrict__ dst,
                              long rows, int sc, int dc) {
  long t = (long)blockIdx.x * blockDim.x + threadIdx.x;
  if (t >= rows * dc) return;
  long r = t / dc;
  int  c = (int)(t - r * dc);
  dst[t] = (c < sc) ? f2bf(src[r * sc + c]) : (unsigned short)0;
}

// ---- C[M,64] = A[M,K](bf16) @ Bswz (bf16) + bias, f32 out. One wave = 16 rows. ----
template <int K>
__global__ void k_gemm_wmma(const unsigned short* __restrict__ A,
                            const unsigned short* __restrict__ Bs,
                            const float* __restrict__ bias,
                            float* __restrict__ C, int M) {
  const int lane = threadIdx.x & 31;
  const int wave = threadIdx.x >> 5;
  const int tile = blockIdx.x * (blockDim.x >> 5) + wave;
  const long row0 = (long)tile * 16;
  if (row0 >= M) return;                     // uniform per wave: EXEC stays all-1 for WMMA
  const int r  = lane & 15;
  const int hi = lane >> 4;
  const long arow = row0 + r;
  const bool aOK = arow < M;
  v8f c0 = {}, c1 = {}, c2 = {}, c3 = {};
#pragma unroll
  for (int kt = 0; kt < (K >> 5); ++kt) {
    ABits a;
    if (aOK) {
      // 16-bit A 16x32 layout: lane<16 -> K {0..7,16..23}; lane>=16 -> K {8..15,24..31}
      const uint4* ap = (const uint4*)(A + arow * K + kt * 32 + hi * 8);
      a.q[0] = ap[0];
      a.q[1] = ap[2];            // +16 cols = +32B
    } else {
      a.q[0] = make_uint4(0u, 0u, 0u, 0u);
      a.q[1] = make_uint4(0u, 0u, 0u, 0u);
    }
    const uint4* bp = (const uint4*)(Bs + ((size_t)(kt * 4) * 32 + lane) * 16);
    ABits b0, b1, b2, b3;                    // each nt stride = 32*16 ushort = 64 uint4
    b0.q[0] = bp[0];   b0.q[1] = bp[1];
    b1.q[0] = bp[64];  b1.q[1] = bp[65];
    b2.q[0] = bp[128]; b2.q[1] = bp[129];
    b3.q[0] = bp[192]; b3.q[1] = bp[193];
    c0 = __builtin_amdgcn_wmma_f32_16x16x32_bf16(false, a.v, false, b0.v, (short)0, c0, false, false);
    c1 = __builtin_amdgcn_wmma_f32_16x16x32_bf16(false, a.v, false, b1.v, (short)0, c1, false, false);
    c2 = __builtin_amdgcn_wmma_f32_16x16x32_bf16(false, a.v, false, b2.v, (short)0, c2, false, false);
    c3 = __builtin_amdgcn_wmma_f32_16x16x32_bf16(false, a.v, false, b3.v, (short)0, c3, false, false);
  }
  // C/D layout: VGPR e -> M = e + (lane>=16 ? 8:0); N = (lane&15) + nt*16
#pragma unroll
  for (int e = 0; e < 8; ++e) {
    long orow = row0 + hi * 8 + e;
    if (orow < M) {
      float* cp = C + orow * 64;
      float z0 = bias ? bias[r]      : 0.f;
      float z1 = bias ? bias[16 + r] : 0.f;
      float z2 = bias ? bias[32 + r] : 0.f;
      float z3 = bias ? bias[48 + r] : 0.f;
      cp[r]      = c0[e] + z0;
      cp[16 + r] = c1[e] + z1;
      cp[32 + r] = c2[e] + z2;
      cp[48 + r] = c3[e] + z3;
    }
  }
}

// ---- alpha_src[n,h] = sum_c H[n,h*16+c] * att[h*16+c] ----
__global__ void k_alpha_src(const float* __restrict__ H, const float* __restrict__ att,
                            float* __restrict__ out, int N) {
  int t = blockIdx.x * blockDim.x + threadIdx.x;
  if (t >= N * 4) return;
  int n = t >> 2, h = t & 3;
  const float* hp = H + (long)n * 64 + h * 16;
  const float* ap = att + h * 16;
  float s = 0.f;
#pragma unroll
  for (int c = 0; c < 16; ++c) s += hp[c] * ap[c];
  out[t] = s;
}

// ---- WA[k,h] = sum_c W[k, h*16+c] * att_dst[h*16+c]  (64x4), single block of 256 ----
__global__ void k_wa(const float* __restrict__ W, const float* __restrict__ attd,
                     float* __restrict__ WA) {
  int t = threadIdx.x;
  int k = t >> 2, h = t & 3;
  float s = 0.f;
#pragma unroll
  for (int c = 0; c < 16; ++c) s += W[k * 64 + h * 16 + c] * attd[h * 16 + c];
  WA[t] = s;
}

// ---- alpha_dst[n,h] = sum_k X[n,k] * WA[k,h] ----
__global__ void k_alpha_dst(const float* __restrict__ X, const float* __restrict__ WA,
                            float* __restrict__ out, int N) {
  int t = blockIdx.x * blockDim.x + threadIdx.x;
  if (t >= N * 4) return;
  int n = t >> 2, h = t & 3;
  const float* xp = X + (long)n * 64;
  float s = 0.f;
#pragma unroll
  for (int k = 0; k < 64; ++k) s += xp[k] * WA[k * 4 + h];
  out[t] = s;
}

__global__ void k_seg_init(unsigned* __restrict__ amax, float* __restrict__ denom, int n4) {
  int t = blockIdx.x * blockDim.x + threadIdx.x;
  if (t >= n4) return;
  amax[t] = fenc(-__builtin_huge_valf());
  denom[t] = 0.f;
}

__device__ __forceinline__ float edge_alpha(const float* as, const float* ad, int s, int d, int h) {
  float a = as[s * 4 + h] + ad[d * 4 + h];
  return (a > 0.f) ? a : 0.2f * a;       // leaky_relu 0.2
}

__global__ void k_edge_max(const int* __restrict__ ei, int E,
                           const float* __restrict__ as, const float* __restrict__ ad,
                           unsigned* __restrict__ amax) {
  int t = blockIdx.x * blockDim.x + threadIdx.x;
  if (t >= E * 4) return;
  int e = t >> 2, h = t & 3;
  int s = ei[e], d = ei[E + e];
  atomicMax(amax + d * 4 + h, fenc(edge_alpha(as, ad, s, d, h)));
}

__global__ void k_edge_sum(const int* __restrict__ ei, int E,
                           const float* __restrict__ as, const float* __restrict__ ad,
                           const unsigned* __restrict__ amax, float* __restrict__ denom) {
  int t = blockIdx.x * blockDim.x + threadIdx.x;
  if (t >= E * 4) return;
  int e = t >> 2, h = t & 3;
  int s = ei[e], d = ei[E + e];
  float a = edge_alpha(as, ad, s, d, h);
  atomicAdd(denom + d * 4 + h, __expf(a - fdec(amax[d * 4 + h])));
}

__global__ void k_edge_msg(const int* __restrict__ ei, int E,
                           const float* __restrict__ as, const float* __restrict__ ad,
                           const unsigned* __restrict__ amax, const float* __restrict__ denom,
                           const float* __restrict__ H, float* __restrict__ acc) {
  long t = (long)blockIdx.x * blockDim.x + threadIdx.x;
  if (t >= (long)E * 64) return;
  int e = (int)(t >> 6), c = (int)(t & 63), h = c >> 4;
  int s = ei[e], d = ei[E + e];
  float a = edge_alpha(as, ad, s, d, h);
  float w = __expf(a - fdec(amax[d * 4 + h])) / (denom[d * 4 + h] + 1e-16f);
  atomicAdd(acc + (long)d * 64 + c, H[(long)s * 64 + c] * w);
}

// acc[n,c] = b0[c] + b1[c] (+ b2[c])
__global__ void k_acc_init(float* __restrict__ acc, long N, const float* __restrict__ b0,
                           const float* __restrict__ b1, const float* __restrict__ b2) {
  long t = (long)blockIdx.x * blockDim.x + threadIdx.x;
  if (t >= N * 64) return;
  int c = (int)(t & 63);
  float v = b0[c] + b1[c];
  if (b2) v += b2[c];
  acc[t] = v;
}

__global__ void k_relu(float* __restrict__ dst, const float* __restrict__ src, long n) {
  long t = (long)blockIdx.x * blockDim.x + threadIdx.x;
  if (t >= n) return;
  dst[t] = fmaxf(src[t], 0.f);
}

// out[n,2] = relu(T[n,:]@w1 + b1) @ w2 + b2
__global__ void k_classifier(const float* __restrict__ T, const float* __restrict__ w1,
                             const float* __restrict__ b1, const float* __restrict__ w2,
                             const float* __restrict__ b2, float* __restrict__ out, int N) {
  int n = blockIdx.x * blockDim.x + threadIdx.x;
  if (n >= N) return;
  const float* tp = T + (long)n * 64;
  float o0 = b2[0], o1 = b2[1];
  for (int j = 0; j < 32; ++j) {
    float h = b1[j];
#pragma unroll
    for (int k = 0; k < 64; ++k) h += tp[k] * w1[k * 32 + j];
    h = fmaxf(h, 0.f);
    o0 += h * w2[j * 2];
    o1 += h * w2[j * 2 + 1];
  }
  out[(long)n * 2]     = o0;
  out[(long)n * 2 + 1] = o1;
}

static inline unsigned nblk(long n, int b) { return (unsigned)((n + b - 1) / b); }

extern "C" void kernel_launch(void* const* d_in, const int* in_sizes, int n_in,
                              void* d_out, int out_size, void* d_ws, size_t ws_size,
                              hipStream_t stream) {
  const float* x_user    = (const float*)d_in[0];
  const float* x_tx      = (const float*)d_in[1];
  const float* x_merch   = (const float*)d_in[2];
  const int*   ei_ut     = (const int*)d_in[3];
  const int*   ei_tu     = (const int*)d_in[4];
  const int*   ei_mt     = (const int*)d_in[5];
  const int*   ei_tm     = (const int*)d_in[6];
  const int*   ei_uu     = (const int*)d_in[7];
  const int*   ei_mm     = (const int*)d_in[8];
  const int*   ei_tt     = (const int*)d_in[9];
  const float* Win_user  = (const float*)d_in[10];
  const float* bin_user  = (const float*)d_in[11];
  const float* Win_tx    = (const float*)d_in[12];
  const float* bin_tx    = (const float*)d_in[13];
  const float* Win_merch = (const float*)d_in[14];
  const float* bin_merch = (const float*)d_in[15];
  const float* lin_w     = (const float*)d_in[16];
  const float* att_src   = (const float*)d_in[17];
  const float* att_dst   = (const float*)d_in[18];
  const float* conv_bias = (const float*)d_in[19];
  const float* cls_w1    = (const float*)d_in[20];
  const float* cls_b1    = (const float*)d_in[21];
  const float* cls_w2    = (const float*)d_in[22];
  const float* cls_b2    = (const float*)d_in[23];
  float* out = (float*)d_out;

  // ---- carve workspace ----
  char* wp = (char*)d_ws;
  auto carve = [&](size_t bytes) -> char* {
    char* p = wp;
    wp += (bytes + 255) & ~(size_t)255;
    return p;
  };
  float* cur_u = (float*)carve((size_t)NUSER * 64 * 4);
  float* cur_t = (float*)carve((size_t)NTX * 64 * 4);
  float* cur_m = (float*)carve((size_t)NMERCH * 64 * 4);
  float* acc_u = (float*)carve((size_t)NUSER * 64 * 4);
  float* acc_t = (float*)carve((size_t)NTX * 64 * 4);
  float* acc_m = (float*)carve((size_t)NMERCH * 64 * 4);
  unsigned short* ub = (unsigned short*)carve((size_t)NUSER * 64 * 2);
  unsigned short* tb = (unsigned short*)carve((size_t)NTX * 64 * 2);
  unsigned short* mb = (unsigned short*)carve((size_t)NMERCH * 64 * 2);
  float* Hbuf  = (float*)carve((size_t)NTX * 64 * 4);
  float* alf_s = (float*)carve((size_t)NTX * 4 * 4);
  float* alf_d = (float*)carve((size_t)NTX * 4 * 4);
  unsigned* amax = (unsigned*)carve((size_t)NTX * 4 * 4);
  float* denom = (float*)carve((size_t)NTX * 4 * 4);
  float* WA    = (float*)carve(64 * 4 * 4);
  unsigned short* wswLin  = (unsigned short*)carve((size_t)14 * 4096 * 2);
  unsigned short* wswTx   = (unsigned short*)carve((size_t)4096 * 2);
  unsigned short* wswUser = (unsigned short*)carve((size_t)2048 * 2);
  unsigned short* wswMrch = (unsigned short*)carve((size_t)2048 * 2);

  // ---- 1. weight swizzle/convert (every call; deterministic) ----
  for (int i = 0; i < 14; ++i)
    k_swizzle_w<<<nblk(64 * 64, 256), 256, 0, stream>>>(lin_w + (size_t)i * 4096, wswLin + (size_t)i * 4096, 64, 64);
  k_swizzle_w<<<nblk(64 * 64, 256), 256, 0, stream>>>(Win_tx, wswTx, 64, 64);
  k_swizzle_w<<<nblk(32 * 64, 256), 256, 0, stream>>>(Win_user, wswUser, 32, 32);
  k_swizzle_w<<<nblk(32 * 64, 256), 256, 0, stream>>>(Win_merch, wswMrch, 16, 32);  // pad K 16->32

  // ---- 2. input features -> bf16 (merch padded to K=32) ----
  k_f32_to_bf16<<<nblk((long)NTX * 64, 256), 256, 0, stream>>>(x_tx, tb, NTX, 64, 64);
  k_f32_to_bf16<<<nblk((long)NUSER * 32, 256), 256, 0, stream>>>(x_user, ub, NUSER, 32, 32);
  k_f32_to_bf16<<<nblk((long)NMERCH * 32, 256), 256, 0, stream>>>(x_merch, mb, NMERCH, 16, 32);

  // ---- 3. input projections (WMMA) ----
  k_gemm_wmma<64><<<nblk(NTX, 64), 128, 0, stream>>>(tb, wswTx, bin_tx, cur_t, NTX);
  k_gemm_wmma<32><<<nblk(NUSER, 64), 128, 0, stream>>>(ub, wswUser, bin_user, cur_u, NUSER);
  k_gemm_wmma<32><<<nblk(NMERCH, 64), 128, 0, stream>>>(mb, wswMrch, bin_merch, cur_m, NMERCH);

  // ---- conv descriptors ----
  const int* eis[7]  = {ei_ut, ei_tu, ei_mt, ei_tm, ei_uu, ei_mm, ei_tt};
  const int  Es[7]   = {300000, 300000, 300000, 300000, 500000, 100000, 600000};
  const int  srcT[7] = {0, 1, 2, 1, 0, 2, 1};   // 0=user 1=tx 2=merch
  const int  dstT[7] = {1, 0, 1, 2, 0, 2, 1};
  const long Nn[3]   = {NUSER, NTX, NMERCH};
  unsigned short* bfB[3] = {ub, tb, mb};
  float* curF[3] = {cur_u, cur_t, cur_m};
  float* accF[3] = {acc_u, acc_t, acc_m};

  for (int l = 0; l < 2; ++l) {
    // bf16 snapshots of current features (K=64 now for all types)
    k_f32_to_bf16<<<nblk((long)NUSER * 64, 256), 256, 0, stream>>>(cur_u, ub, NUSER, 64, 64);
    k_f32_to_bf16<<<nblk((long)NTX * 64, 256), 256, 0, stream>>>(cur_t, tb, NTX, 64, 64);
    k_f32_to_bf16<<<nblk((long)NMERCH * 64, 256), 256, 0, stream>>>(cur_m, mb, NMERCH, 64, 64);

    // accumulators start at summed conv biases per destination type
    const float* B = conv_bias + (size_t)l * 7 * 64;
    k_acc_init<<<nblk((long)NTX * 64, 256), 256, 0, stream>>>(acc_t, NTX, B + 0 * 64, B + 2 * 64, B + 6 * 64);
    k_acc_init<<<nblk((long)NUSER * 64, 256), 256, 0, stream>>>(acc_u, NUSER, B + 1 * 64, B + 4 * 64, nullptr);
    k_acc_init<<<nblk((long)NMERCH * 64, 256), 256, 0, stream>>>(acc_m, NMERCH, B + 3 * 64, B + 5 * 64, nullptr);

    for (int i = 0; i < 7; ++i) {
      const int st = srcT[i], dt = dstT[i];
      const long Ns = Nn[st], Nd = Nn[dt];
      const int E = Es[i];
      const float* W    = lin_w + ((size_t)l * 7 + i) * 4096;
      const float* aS   = att_src + ((size_t)l * 7 + i) * 64;
      const float* aD   = att_dst + ((size_t)l * 7 + i) * 64;
      const unsigned short* Wsw = wswLin + ((size_t)l * 7 + i) * 4096;

      // h_src = x_src @ W (WMMA)
      k_gemm_wmma<64><<<nblk(Ns, 64), 128, 0, stream>>>(bfB[st], Wsw, nullptr, Hbuf, (int)Ns);
      // attention logits
      k_alpha_src<<<nblk(Ns * 4, 256), 256, 0, stream>>>(Hbuf, aS, alf_s, (int)Ns);
      k_wa<<<1, 256, 0, stream>>>(W, aD, WA);
      k_alpha_dst<<<nblk(Nd * 4, 256), 256, 0, stream>>>(curF[dt], WA, alf_d, (int)Nd);
      // segment softmax + scatter
      k_seg_init<<<nblk(Nd * 4, 256), 256, 0, stream>>>(amax, denom, (int)(Nd * 4));
      k_edge_max<<<nblk((long)E * 4, 256), 256, 0, stream>>>(eis[i], E, alf_s, alf_d, amax);
      k_edge_sum<<<nblk((long)E * 4, 256), 256, 0, stream>>>(eis[i], E, alf_s, alf_d, amax, denom);
      k_edge_msg<<<nblk((long)E * 64, 256), 256, 0, stream>>>(eis[i], E, alf_s, alf_d, amax, denom, Hbuf, accF[dt]);
    }

    k_relu<<<nblk((long)NUSER * 64, 256), 256, 0, stream>>>(cur_u, acc_u, (long)NUSER * 64);
    k_relu<<<nblk((long)NTX * 64, 256), 256, 0, stream>>>(cur_t, acc_t, (long)NTX * 64);
    k_relu<<<nblk((long)NMERCH * 64, 256), 256, 0, stream>>>(cur_m, acc_m, (long)NMERCH * 64);
  }

  // ---- classifier on tx nodes ----
  k_classifier<<<nblk(NTX, 128), 128, 0, stream>>>(cur_t, cls_w1, cls_b1, cls_w2, cls_b2, out, NTX);
}
